// RQSplineCouplingLayer_86096914415940
// MI455X (gfx1250) — matrix operand order repaired and checked by reading, not compiled, not run on hardware
//
#include <hip/hip_runtime.h>
#include <hip/hip_bf16.h>
#include <cstdint>

// ---------------------------------------------------------------------------
// RQ-spline coupling layer for MI455X (gfx1250, wave32).
// conv2 hotspot on the WMMA pipe (v_wmma_f32_16x16x32_f16), halo tiles moved
// with GLOBAL_LOAD_ASYNC_TO_LDS_B128 (ASYNCcnt), weights prepacked once and
// streamed from L2 straight into B-operand VGPRs.
// ---------------------------------------------------------------------------

typedef _Float16 v16h __attribute__((ext_vector_type(16)));
typedef _Float16 v8h  __attribute__((ext_vector_type(8)));
typedef float    v8f  __attribute__((ext_vector_type(8)));

#define LSZ   128
#define LL    (128 * 128)
#define NB    64
#define HID   32
#define COUT  23
#define NSEG  8
#define BOUND 5.0f

// ---------------------------------------------------------------------------
// Kernel 1: conv1 (1 -> 32, 3x3 circular) + tanh, output fp16 [b][y][x][c]
// ---------------------------------------------------------------------------
__global__ __launch_bounds__(256) void conv1_tanh_kernel(
    const float* __restrict__ x, const float* __restrict__ W,
    const float* __restrict__ bias, _Float16* __restrict__ h)
{
    __shared__ float sW[HID * 9];
    __shared__ float sb[HID];
    const int t = threadIdx.x;
    for (int i = t; i < HID * 9; i += 256) sW[i] = W[i];
    if (t < HID) sb[t] = bias[t];
    __syncthreads();

    const int gid = blockIdx.x * 256 + t;          // one pixel per thread
    const int b   = gid >> 14;
    const int rem = gid & (LL - 1);
    const int y   = rem >> 7;
    const int xx  = rem & (LSZ - 1);
    const float* img = x + (size_t)b * LL;

    float v[9];
#pragma unroll
    for (int dy = 0; dy < 3; ++dy) {
        const int yy = (y + dy + LSZ - 1) & (LSZ - 1);
#pragma unroll
        for (int dx = 0; dx < 3; ++dx) {
            const int xw = (xx + dx + LSZ - 1) & (LSZ - 1);
            v[dy * 3 + dx] = img[yy * LSZ + xw];
        }
    }

    _Float16 hc[HID];
#pragma unroll
    for (int c = 0; c < HID; ++c) {
        float s = sb[c];
#pragma unroll
        for (int k = 0; k < 9; ++k) s = fmaf(sW[c * 9 + k], v[k], s);
        hc[c] = (_Float16)tanhf(s);
    }
    uint4*       dst = (uint4*)(h + (size_t)gid * HID);
    const uint4* src = (const uint4*)hc;
    dst[0] = src[0]; dst[1] = src[1]; dst[2] = src[2]; dst[3] = src[3];
}

// ---------------------------------------------------------------------------
// Prepack W2 into the WMMA B-operand layout this kernel uses (f16, contiguous k), done ONCE per conv:
// pk[((nt*9+tap)*32+lane)*16 + j],  lane = col N (+16*nt), K = j+16*(lane>=16)
// ---------------------------------------------------------------------------
__global__ __launch_bounds__(256) void pack_w2_kernel(
    const float* __restrict__ W2, _Float16* __restrict__ pk)
{
    const int i = blockIdx.x * 256 + threadIdx.x;   // 9216 entries
    if (i >= 2 * 9 * 32 * 16) return;
    const int j    = i & 15;
    const int lane = (i >> 4) & 31;
    const int tap  = (i >> 9) % 9;
    const int nt   = i / (9 * 512);
    const int k    = j + ((lane >> 4) << 4);        // input channel 0..31
    const int n    = (lane & 15) + (nt << 4);       // output channel
    const float wv = (n < COUT)
        ? W2[((n * HID + k) * 3 + tap / 3) * 3 + (tap % 3)] : 0.0f;
    pk[i] = (_Float16)wv;
}

// ---------------------------------------------------------------------------
// Kernel 2: conv2 (32 -> 23, 3x3 circular) via v_wmma_f32_16x16x32_f16,
// fused with the RQ spline + ldj reduction.
// Block = (batch b, row-pair yp): rows 2yp, 2yp+1. 8 waves; wave w owns
// row (w>>2), pixels [(w&3)*32, +32) as two 16-pixel M-tiles.
// ---------------------------------------------------------------------------
__global__ __launch_bounds__(256) void conv2_spline_kernel(
    const _Float16* __restrict__ h, const _Float16* __restrict__ pk,
    const float* __restrict__ b2, const float* __restrict__ xin,
    float* __restrict__ xout, float* __restrict__ ldj)
{
    __shared__ _Float16 sH[4 * 130 * 32];      // halo: 4 rows x 130 px x 32 ch
    __shared__ float    sP[8][32][33];         // per-wave params (+1 pad)
    __shared__ float    sLdj[8];

    const int t  = threadIdx.x;
    const int yp = blockIdx.x;                 // row pair
    const int b  = blockIdx.y;

    // ---- async halo fill: GLOBAL_LOAD_ASYNC_TO_LDS_B128 (ASYNCcnt path) ----
    {
        const unsigned lds_base = (unsigned)(size_t)(&sH[0]);
        for (int i = t; i < 4 * 130 * 4; i += 256) {
            const int sub = i & 3;
            const int px  = (i >> 2) % 130;
            const int r   = i / 520;
            const int gy  = (2 * yp + r + LSZ - 1) & (LSZ - 1);
            const int gx  = (px + LSZ - 1) & (LSZ - 1);
            const uint64_t ga = (uint64_t)(size_t)(h +
                (((size_t)(b * LSZ + gy) * LSZ + gx) * HID + sub * 8));
            const unsigned la = lds_base + (unsigned)i * 16u;
            asm volatile("global_load_async_to_lds_b128 %0, %1, off"
                         :: "v"(la), "v"(ga) : "memory");
        }
        asm volatile("s_wait_asynccnt 0x0" ::: "memory");
    }
    __syncthreads();

    const int wave   = t >> 5;
    const int lane   = t & 31;
    const int laneHi = lane >> 4;
    const int laneLo = lane & 15;
    const int rsel   = wave >> 2;              // which of the 2 rows
    const int xbase  = (wave & 3) * 32;        // 32-pixel strip

    // ---- 9 taps x (2 M-tiles x 2 N-tiles) WMMAs; B streamed from L2 ----
    v8f acc00 = {}, acc01 = {}, acc10 = {}, acc11 = {};
#pragma unroll
    for (int tap = 0; tap < 9; ++tap) {
        const int ry  = tap / 3;
        const int kx2 = tap % 3;
        const int hrow = rsel + ry;
        const int px0  = xbase + laneLo + kx2;           // halo x, M-tile 0
        const _Float16* base0 = &sH[(hrow * 130 + px0) * HID];
        const _Float16* base1 = base0 + 16 * HID;        // M-tile 1 (+16 px)
        // A layout: half j -> K = j + 16*(lane>=16)
        v8h a0lo = *(const v8h*)(base0 + 16 * laneHi);
        v8h a0hi = *(const v8h*)(base0 + 8 + 16 * laneHi);
        v8h a1lo = *(const v8h*)(base1 + 16 * laneHi);
        v8h a1hi = *(const v8h*)(base1 + 8 + 16 * laneHi);
        v16h A0, A1;
#pragma unroll
        for (int q = 0; q < 8; ++q) {
            A0[q] = a0lo[q]; A0[8 + q] = a0hi[q];
            A1[q] = a1lo[q]; A1[8 + q] = a1hi[q];
        }
        v16h B0 = *(const v16h*)&pk[((0 * 9 + tap) * 32 + lane) * 16];
        v16h B1 = *(const v16h*)&pk[((1 * 9 + tap) * 32 + lane) * 16];
        acc00 = __builtin_amdgcn_wmma_f32_16x16x32_f16(false, A0, false, B0,
                                                       (short)0, acc00, false, false);
        acc01 = __builtin_amdgcn_wmma_f32_16x16x32_f16(false, A0, false, B1,
                                                       (short)0, acc01, false, false);
        acc10 = __builtin_amdgcn_wmma_f32_16x16x32_f16(false, A1, false, B0,
                                                       (short)0, acc10, false, false);
        acc11 = __builtin_amdgcn_wmma_f32_16x16x32_f16(false, A1, false, B1,
                                                       (short)0, acc11, false, false);
    }

    // ---- transpose D through LDS (lane=col N, VGPR v = row M=v+8*laneHi) ----
    const float blo = b2[laneLo];
    const float bhi = (laneLo + 16 < COUT) ? b2[laneLo + 16] : 0.0f;
#pragma unroll
    for (int v = 0; v < 8; ++v) {
        const int M = v + 8 * laneHi;
        sP[wave][M][laneLo]           = acc00[v] + blo;
        sP[wave][M][laneLo + 16]      = acc01[v] + bhi;
        sP[wave][M + 16][laneLo]      = acc10[v] + blo;
        sP[wave][M + 16][laneLo + 16] = acc11[v] + bhi;
    }
    __syncthreads();

    // ---- RQ spline: all 32 lanes, pixel x = xbase + lane of row 2yp+rsel ----
    const int   row = 2 * yp + rsel;
    const size_t oidx = (size_t)b * LL + row * LSZ + xbase + lane;
    const float x0 = xin[oidx];
    const float* P = sP[wave][lane];

    float wr[NSEG], hr[NSEG], dr[NSEG - 1];
#pragma unroll
    for (int i = 0; i < NSEG; ++i)     { wr[i] = P[i]; hr[i] = P[NSEG + i]; }
#pragma unroll
    for (int i = 0; i < NSEG - 1; ++i) dr[i] = P[2 * NSEG + i];

    float mw = wr[0], mh = hr[0];
#pragma unroll
    for (int i = 1; i < NSEG; ++i) { mw = fmaxf(mw, wr[i]); mh = fmaxf(mh, hr[i]); }
    float sw = 0.0f, sh = 0.0f, ew[NSEG], eh[NSEG];
#pragma unroll
    for (int i = 0; i < NSEG; ++i) {
        ew[i] = expf(wr[i] - mw); sw += ew[i];
        eh[i] = expf(hr[i] - mh); sh += eh[i];
    }
    const float fw = (2.0f * BOUND) / sw, fh = (2.0f * BOUND) / sh;
    float widths[NSEG], heights[NSEG];
#pragma unroll
    for (int i = 0; i < NSEG; ++i) { widths[i] = ew[i] * fw; heights[i] = eh[i] * fh; }

    float dv[NSEG + 1];
    dv[0] = 1.0f; dv[NSEG] = 1.0f;
#pragma unroll
    for (int i = 0; i < NSEG - 1; ++i) {
        const float d  = dr[i];
        const float sp = (d > 20.0f) ? d : log1pf(expf(d));     // softplus
        dv[i + 1] = sp + 0.001f;
    }

    const bool  inside = (x0 > -BOUND) && (x0 < BOUND);
    const float xc = fminf(fmaxf(x0, -BOUND), BOUND);

    // segment select: unrolled running select (no scratch from dyn-indexing)
    float cx = -BOUND, cy = -BOUND;
    float xk = -BOUND, yk = -BOUND;
    float wk = widths[0], hk = heights[0], d0 = dv[0], d1 = dv[1];
#pragma unroll
    for (int i = 1; i < NSEG; ++i) {
        cx += widths[i - 1]; cy += heights[i - 1];
        const bool take = (cx <= xc);
        xk = take ? cx : xk;           yk = take ? cy : yk;
        wk = take ? widths[i] : wk;    hk = take ? heights[i] : hk;
        d0 = take ? dv[i] : d0;        d1 = take ? dv[i + 1] : d1;
    }

    const float s   = hk / wk;
    const float tt  = fminf(fmaxf((xc - xk) / wk, 0.0f), 1.0f);
    const float ttc = tt * (1.0f - tt);
    const float den = s + (d1 + d0 - 2.0f * s) * ttc;
    const float yv  = yk + hk * (s * tt * tt + d0 * ttc) / den;
    const float omt = 1.0f - tt;
    const float g   = s * s * (d1 * tt * tt + 2.0f * s * ttc + d0 * omt * omt)
                      / (den * den);

    xout[oidx]  = inside ? yv : x0;
    float ldjv  = inside ? logf(g) : 0.0f;

    // ---- ldj: wave shuffle-reduce, then one atomicAdd per block ----
#pragma unroll
    for (int off = 16; off > 0; off >>= 1) ldjv += __shfl_xor(ldjv, off, 32);
    if (lane == 0) sLdj[wave] = ldjv;
    __syncthreads();
    if (t == 0) {
        float sum = 0.0f;
#pragma unroll
        for (int wv = 0; wv < 8; ++wv) sum += sLdj[wv];
        atomicAdd(&ldj[b], sum);
    }
}

__global__ void zero_ldj_kernel(float* __restrict__ p) { p[threadIdx.x] = 0.0f; }

// ---------------------------------------------------------------------------
extern "C" void kernel_launch(void* const* d_in, const int* in_sizes, int n_in,
                              void* d_out, int out_size, void* d_ws, size_t ws_size,
                              hipStream_t stream) {
    (void)in_sizes; (void)n_in; (void)out_size; (void)ws_size;
    const float* phi_a = (const float*)d_in[0];
    const float* phi_b = (const float*)d_in[1];
    const float* Wa1   = (const float*)d_in[2];
    const float* ba1   = (const float*)d_in[3];
    const float* Wa2   = (const float*)d_in[4];
    const float* ba2   = (const float*)d_in[5];
    const float* Wb1   = (const float*)d_in[6];
    const float* bb1   = (const float*)d_in[7];
    const float* Wb2   = (const float*)d_in[8];
    const float* bb2   = (const float*)d_in[9];

    float* xa  = (float*)d_out;                 // (64,1,128,128)
    float* xb  = xa + (size_t)NB * LL;          // (64,1,128,128)
    float* ldj = xb + (size_t)NB * LL;          // (64,)

    _Float16* hbuf = (_Float16*)d_ws;                               // 64 MB
    _Float16* wpk  = (_Float16*)((char*)d_ws + (size_t)NB * LL * HID * 2);

    const dim3 rowGrid(LSZ / 2, NB);            // row-pairs x batch
    const int  packBlocks = (2 * 9 * 32 * 16 + 255) / 256;

    zero_ldj_kernel<<<1, NB, 0, stream>>>(ldj);
    // coupling A: params from phi_b, transform phi_a -> xa
    pack_w2_kernel<<<packBlocks, 256, 0, stream>>>(Wb2, wpk);
    conv1_tanh_kernel<<<(NB * LL) / 256, 256, 0, stream>>>(phi_b, Wb1, bb1, hbuf);
    conv2_spline_kernel<<<rowGrid, 256, 0, stream>>>(hbuf, wpk, bb2, phi_a, xa, ldj);
    // coupling B: params from xa, transform phi_b -> xb
    pack_w2_kernel<<<packBlocks, 256, 0, stream>>>(Wa2, wpk);
    conv1_tanh_kernel<<<(NB * LL) / 256, 256, 0, stream>>>(xa, Wa1, ba1, hbuf);
    conv2_spline_kernel<<<rowGrid, 256, 0, stream>>>(hbuf, wpk, ba2, phi_b, xb, ldj);
}
